// MoE_38336878084277
// MI455X (gfx1250) — matrix operand order repaired and checked by reading, not compiled, not run on hardware
//
#include <hip/hip_runtime.h>
#include <math.h>

// ---- problem constants ----
#define SB   8
#define TT   2048
#define SS   (SB*TT)      // 16384 tokens
#define CC   1024
#define EE   8
#define KK   2
#define HH   4096

// ---- tiling ----
#define BM    64          // tokens per block
#define TMAX  (SS/BM)     // 256 max tiles per expert
#define HC    128         // H chunk
#define KS    32          // K step (= WMMA K for bf16)
#define NHC   (HH/HC)     // 32
#define NKS   (CC/KS)     // 32
#define NK2   (HC/KS)     // 4

typedef __attribute__((ext_vector_type(16))) __bf16 v16bf;
typedef __attribute__((ext_vector_type(8)))  float  v8f;
typedef __attribute__((ext_vector_type(8)))  __bf16 bf16x8;
typedef __attribute__((ext_vector_type(4)))  __bf16 bf16x4;

// ---------------- init ----------------
__global__ void moe_init(float* y, int* counts, float* fcnt, float* psum) {
    size_t i = (size_t)blockIdx.x * blockDim.x + threadIdx.x;
    size_t n4 = (size_t)SS * CC / 4;
    float4 z = make_float4(0.f, 0.f, 0.f, 0.f);
    for (size_t j = i; j < n4; j += (size_t)gridDim.x * blockDim.x)
        ((float4*)y)[j] = z;
    if (blockIdx.x == 0 && threadIdx.x < EE) {
        counts[threadIdx.x] = 0;
        fcnt[threadIdx.x] = 0.f;
        psum[threadIdx.x] = 0.f;
    }
}

// ---------------- cast x -> bf16 ----------------
__global__ void cast_x(const float* __restrict__ x, __bf16* __restrict__ xbf) {
    size_t i = (size_t)blockIdx.x * blockDim.x + threadIdx.x;
    size_t n4 = (size_t)SS * CC / 4;
    for (size_t j = i; j < n4; j += (size_t)gridDim.x * blockDim.x) {
        float4 v = ((const float4*)x)[j];
        bf16x4 o;
        o[0] = (__bf16)v.x; o[1] = (__bf16)v.y; o[2] = (__bf16)v.z; o[3] = (__bf16)v.w;
        ((bf16x4*)xbf)[j] = o;
    }
}

// ---------------- weight pre-swizzle: fp32 -> bf16, B-fragment tile order ----------------
// W1/W3 (C x H per expert) -> slabs [e][hc][ks][n(128)][k(32)], slab = 8KB contiguous
__global__ void swizzle_w13(const float* __restrict__ src, __bf16* __restrict__ dst) {
    size_t chunk = (size_t)blockIdx.x * blockDim.x + threadIdx.x;   // 4 elems along H
    size_t base = chunk * 4;
    if (base >= (size_t)EE * CC * HH) return;
    int h = (int)(base % HH);
    int c = (int)((base / HH) % CC);
    int e = (int)(base / ((size_t)HH * CC));
    float4 v = *(const float4*)(src + base);
    int hc = h >> 7, n0 = h & 127, ks = c >> 5, k = c & 31;
    size_t db = ((((size_t)e * NHC + hc) * NKS + ks) * (HC * KS)) + k;
    dst[db + (size_t)(n0 + 0) * KS] = (__bf16)v.x;
    dst[db + (size_t)(n0 + 1) * KS] = (__bf16)v.y;
    dst[db + (size_t)(n0 + 2) * KS] = (__bf16)v.z;
    dst[db + (size_t)(n0 + 3) * KS] = (__bf16)v.w;
}
// W2 (H x C per expert) -> slabs [e][hc][kk][n(1024)][k(32)], slab = 64KB contiguous
__global__ void swizzle_w2(const float* __restrict__ src, __bf16* __restrict__ dst) {
    size_t chunk = (size_t)blockIdx.x * blockDim.x + threadIdx.x;   // 4 elems along C
    size_t base = chunk * 4;
    if (base >= (size_t)EE * HH * CC) return;
    int n = (int)(base % CC);
    int hh = (int)((base / CC) % HH);
    int e = (int)(base / ((size_t)CC * HH));
    float4 v = *(const float4*)(src + base);
    int hc = hh >> 7, kk = (hh >> 5) & 3, k = hh & 31;
    size_t db = ((((size_t)e * NHC + hc) * NK2 + kk) * (CC * KS)) + k;
    dst[db + (size_t)(n + 0) * KS] = (__bf16)v.x;
    dst[db + (size_t)(n + 1) * KS] = (__bf16)v.y;
    dst[db + (size_t)(n + 2) * KS] = (__bf16)v.z;
    dst[db + (size_t)(n + 3) * KS] = (__bf16)v.w;
}

// ---------------- gate / router ----------------
__launch_bounds__(256)
__global__ void gate_kernel(const float* __restrict__ x, const float* __restrict__ Wg,
                            const float* __restrict__ bg,
                            int* counts, int* bucket, float* wts,
                            float* fcnt, float* psum) {
    int wave = threadIdx.x >> 5;
    int lane = threadIdx.x & 31;
    int tok = blockIdx.x * 8 + wave;
    if (tok >= SS) return;
    float acc[EE];
#pragma unroll
    for (int e = 0; e < EE; ++e) acc[e] = 0.f;
    const float* xr = x + (size_t)tok * CC;
    for (int c = lane; c < CC; c += 32) {
        float xv = xr[c];
        const float* wr = Wg + (size_t)c * EE;
#pragma unroll
        for (int e = 0; e < EE; ++e) acc[e] += xv * wr[e];
    }
#pragma unroll
    for (int e = 0; e < EE; ++e) {
#pragma unroll
        for (int off = 16; off > 0; off >>= 1)
            acc[e] += __shfl_xor(acc[e], off, 32);
    }
    if (lane == 0) {
        float lg[EE], p[EE];
        float mx = -1e30f;
#pragma unroll
        for (int e = 0; e < EE; ++e) { lg[e] = acc[e] + bg[e]; mx = fmaxf(mx, lg[e]); }
        float se = 0.f;
#pragma unroll
        for (int e = 0; e < EE; ++e) { p[e] = __expf(lg[e] - mx); se += p[e]; }
        float inv = 1.f / se;
#pragma unroll
        for (int e = 0; e < EE; ++e) p[e] *= inv;
        int i0 = 0;
#pragma unroll
        for (int e = 1; e < EE; ++e) if (p[e] > p[i0]) i0 = e;
        int i1 = (i0 == 0) ? 1 : 0;
#pragma unroll
        for (int e = 0; e < EE; ++e) if (e != i0 && p[e] > p[i1]) i1 = e;
        float wsum = p[i0] + p[i1];
        float c0 = p[i0] / wsum, c1 = p[i1] / wsum;
        int pos0 = atomicAdd(&counts[i0], 1);
        bucket[i0 * SS + pos0] = tok; wts[i0 * SS + pos0] = c0;
        int pos1 = atomicAdd(&counts[i1], 1);
        bucket[i1 * SS + pos1] = tok; wts[i1 * SS + pos1] = c1;
        atomicAdd(&fcnt[i0], 1.f);
        atomicAdd(&fcnt[i1], 1.f);
#pragma unroll
        for (int e = 0; e < EE; ++e) atomicAdd(&psum[e], p[e]);
    }
}

// ---------------- fragment loaders (CDNA5 documented layouts) ----------------
// A 16x32 bf16 row-major in LDS
__device__ inline v16bf ld_frag_a(const __bf16* base, int ld, int lane) {
    int half = lane >> 4, m = lane & 15;
    union { uint4 q[2]; v16bf v; } f;
    f.q[0] = *(const uint4*)(base + m * ld + 8 * half);
    f.q[1] = *(const uint4*)(base + m * ld + 16 + 8 * half);
    return f.v;
}
// B 32x16 bf16 from [n][k]-ordered LDS; baseT points at row n0
__device__ inline v16bf ld_frag_bt(const __bf16* baseT, int lane) {
    int half = lane >> 4, n = lane & 15;
    union { uint4 q[2]; v16bf v; } f;
    const __bf16* p = baseT + n * KS + 8 * half;
    f.q[0] = *(const uint4*)(p);
    f.q[1] = *(const uint4*)(p + 16);
    return f.v;
}

// async 16B copy global -> LDS (ASYNCcnt-tracked, bypasses VGPRs)
__device__ inline void async_g2l_b128(void* lds_ptr, const void* gptr) {
    unsigned lds_off = (unsigned)(uintptr_t)lds_ptr;   // low 32 bits = LDS byte offset
    asm volatile("global_load_async_to_lds_b128 %0, %1, off"
                 :: "v"(lds_off), "v"(gptr) : "memory");
}
__device__ inline void wait_async_0()  { asm volatile("s_wait_asynccnt 0x0"  ::: "memory"); }
__device__ inline void wait_async_4()  { asm volatile("s_wait_asynccnt 0x4"  ::: "memory"); }
__device__ inline void wait_async_16() { asm volatile("s_wait_asynccnt 0x10" ::: "memory"); }

// issue one W1+W3 slab (2 x 8KB) as async copies: 4 x 16B per thread
__device__ inline void issue_w13(const __bf16* W1s, const __bf16* W3s, size_t slab,
                                 __bf16* dst0, __bf16* dst1, int tid) {
#pragma unroll
    for (int j = 0; j < 4; ++j) {
        int q = j * 256 + tid;          // 0..1023 chunks of 8 bf16
        int mat = q >> 9;
        int off = (q & 511) * 8;
        const __bf16* src = (mat ? W3s : W1s) + slab + off;
        __bf16* d = (mat ? dst1 : dst0) + off;
        async_g2l_b128(d, src);
    }
}
// issue one W2 slab (64KB): 16 x 16B per thread
__device__ inline void issue_w2(const __bf16* W2s, size_t slab, __bf16* dst, int tid) {
#pragma unroll
    for (int j = 0; j < 16; ++j) {
        int off = (j * 256 + tid) * 8;
        async_g2l_b128(dst + off, W2s + slab + off);
    }
}

// ---------------- fused per-expert SwiGLU MLP ----------------
__launch_bounds__(256, 1)
__global__ void moe_ffn(const __bf16* __restrict__ xbf,
                        const __bf16* __restrict__ W1s, const float* __restrict__ b1,
                        const __bf16* __restrict__ W3s, const float* __restrict__ b3,
                        const __bf16* __restrict__ W2s, const float* __restrict__ b2,
                        const int* __restrict__ counts, const int* __restrict__ bucket,
                        const float* __restrict__ wts, float* __restrict__ y) {
    const int e = blockIdx.x >> 8;          // expert
    const int t = blockIdx.x & (TMAX - 1);  // tile within expert
    const int cnt = counts[e];
    const int row0 = t * BM;
    if (row0 >= cnt) return;
    const int rows = (cnt - row0 < BM) ? (cnt - row0) : BM;

    __shared__ __align__(16) __bf16 xs[BM][CC];           // 128 KB (A operand)
    __shared__ __align__(16) __bf16 w13d[2][2][HC * KS];  // 32 KB, double-buffered [n][k]
    __shared__ __align__(16) __bf16 hs[BM][HC];           // 16 KB (A operand)
    __shared__ __align__(16) __bf16 w2d[2][KS * CC];      // 128 KB, double-buffered [n][k]
    __shared__ int   tokid[BM];
    __shared__ float cew[BM];

    const int tid = threadIdx.x;
    const int lane = tid & 31;
    const int w = tid >> 5;                 // wave id 0..7
    const int half = lane >> 4;

    if (tid < BM) {
        int r = tid;
        if (r < rows) {
            tokid[r] = bucket[e * SS + row0 + r];
            cew[r]   = wts[e * SS + row0 + r];
        } else { tokid[r] = 0; cew[r] = 0.f; }
    }
    __syncthreads();

    // stage x tile: async global->LDS (bf16); zero-pad tail rows
    for (int i = tid; i < BM * (CC / 8); i += 256) {
        int r = i / (CC / 8);
        int c8 = i % (CC / 8);
        if (r < rows) {
            async_g2l_b128(&xs[r][c8 * 8],
                           xbf + (size_t)tokid[r] * CC + (size_t)c8 * 8);
        } else {
            *((bf16x8*)&xs[r][c8 * 8]) = (bf16x8)(__bf16)0.0f;
        }
    }
    wait_async_0();
    __syncthreads();

    v8f yacc[32];   // 4 M-tiles x 8 N-tiles of 16x16 f32, N range = [w*128, w*128+128)
#pragma unroll
    for (int i = 0; i < 32; ++i) yacc[i] = (v8f)0.f;

    for (int hc = 0; hc < NHC; ++hc) {
        // ---- GEMM 1+3: a1 = x@W1c, a3 = x@W3c (wave w covers chunk cols [w*16, w*16+16))
        const size_t slabb = (((size_t)e * NHC + hc) * NKS) * (HC * KS);
        issue_w13(W1s, W3s, slabb, &w13d[0][0][0], &w13d[0][1][0], tid);

        v8f a1[4], a3[4];
#pragma unroll
        for (int mt = 0; mt < 4; ++mt) { a1[mt] = (v8f)0.f; a3[mt] = (v8f)0.f; }

        for (int ks = 0; ks < NKS; ++ks) {
            int p = ks & 1;
            if (ks + 1 < NKS) {
                issue_w13(W1s, W3s, slabb + (size_t)(ks + 1) * (HC * KS),
                          &w13d[1 - p][0][0], &w13d[1 - p][1][0], tid);
                wait_async_4();        // previous slab landed (in-order completion)
            } else {
                wait_async_0();
            }
            __syncthreads();
            v16bf bf1 = ld_frag_bt(&w13d[p][0][(w * 16) * KS], lane);
            v16bf bf3 = ld_frag_bt(&w13d[p][1][(w * 16) * KS], lane);
            v16bf af[4];
#pragma unroll
            for (int mt = 0; mt < 4; ++mt)
                af[mt] = ld_frag_a(&xs[mt * 16][ks * KS], CC, lane);
#pragma unroll
            for (int mt = 0; mt < 4; ++mt) {
                a1[mt] = __builtin_amdgcn_wmma_f32_16x16x32_bf16(
                    false, af[mt], false, bf1, (short)0, a1[mt], false, false);
                a3[mt] = __builtin_amdgcn_wmma_f32_16x16x32_bf16(
                    false, af[mt], false, bf3, (short)0, a3[mt], false, false);
            }
            __syncthreads();
        }

        // prefetch first W2 slab while doing SwiGLU math
        const size_t slab2 = (((size_t)e * NHC + hc) * NK2) * (CC * KS);
        issue_w2(W2s, slab2, &w2d[0][0], tid);

        // ---- SwiGLU: h = silu(a1 + b1) * (a3 + b3) -> LDS bf16
        {
            int nloc = w * 16 + (lane & 15);
            float bb1 = b1[(size_t)e * HH + hc * HC + nloc];
            float bb3 = b3[(size_t)e * HH + hc * HC + nloc];
#pragma unroll
            for (int mt = 0; mt < 4; ++mt) {
#pragma unroll
                for (int v = 0; v < 8; ++v) {
                    float z1 = a1[mt][v] + bb1;
                    float z3 = a3[mt][v] + bb3;
                    float hval = (z1 / (1.f + __expf(-z1))) * z3;
                    hs[mt * 16 + v + 8 * half][nloc] = (__bf16)hval;
                }
            }
        }

        // ---- GEMM 2: y += h_chunk @ W2[hc rows, :]
        for (int kk2 = 0; kk2 < NK2; ++kk2) {
            int p = kk2 & 1;
            if (kk2 + 1 < NK2) {
                issue_w2(W2s, slab2 + (size_t)(kk2 + 1) * (CC * KS), &w2d[1 - p][0], tid);
                wait_async_16();
            } else {
                wait_async_0();
            }
            __syncthreads();   // w2 slab ready + hs visible
            v16bf am[4];
#pragma unroll
            for (int mt = 0; mt < 4; ++mt)
                am[mt] = ld_frag_a(&hs[mt * 16][kk2 * KS], HC, lane);
#pragma unroll
            for (int j = 0; j < 8; ++j) {
                v16bf bf = ld_frag_bt(&w2d[p][(w * 128 + j * 16) * KS], lane);
#pragma unroll
                for (int mt = 0; mt < 4; ++mt)
                    yacc[mt * 8 + j] = __builtin_amdgcn_wmma_f32_16x16x32_bf16(
                        false, am[mt], false, bf, (short)0, yacc[mt * 8 + j], false, false);
            }
            __syncthreads();
        }
    }

    // ---- epilogue: y[tok, n] += ce * (acc + b2[n])
#pragma unroll
    for (int mt = 0; mt < 4; ++mt) {
#pragma unroll
        for (int j = 0; j < 8; ++j) {
            int nglob = w * 128 + j * 16 + (lane & 15);
            float b2v = b2[(size_t)e * CC + nglob];
#pragma unroll
            for (int v = 0; v < 8; ++v) {
                int r = mt * 16 + v + 8 * half;
                if (r < rows) {
                    float val = yacc[mt * 8 + j][v] + b2v;
                    atomicAdd(&y[(size_t)tokid[r] * CC + nglob], cew[r] * val);
                }
            }
        }
    }
}

// ---------------- aux loss ----------------
__global__ void aux_finalize(const float* fcnt, const float* psum, float* out_aux) {
    if (threadIdx.x == 0 && blockIdx.x == 0) {
        float aux = 0.f;
        for (int e = 0; e < EE; ++e)
            aux += (fcnt[e] / (float)(SS * KK)) * (psum[e] / (float)SS);
        *out_aux = (float)EE * aux;
    }
}

// ---------------- launcher ----------------
extern "C" void kernel_launch(void* const* d_in, const int* in_sizes, int n_in,
                              void* d_out, int out_size, void* d_ws, size_t ws_size,
                              hipStream_t stream) {
    const float* x  = (const float*)d_in[0];
    const float* Wg = (const float*)d_in[1];
    const float* bg = (const float*)d_in[2];
    const float* W1 = (const float*)d_in[3];
    const float* b1 = (const float*)d_in[4];
    const float* W3 = (const float*)d_in[5];
    const float* b3 = (const float*)d_in[6];
    const float* W2 = (const float*)d_in[7];
    const float* b2 = (const float*)d_in[8];
    float* y = (float*)d_out;                    // [SS*CC] then aux at [SS*CC]

    char* ws = (char*)d_ws;
    size_t off = 0;
    int*    counts = (int*)(ws + off);   off += 256;
    float*  fcnt   = (float*)(ws + off); off += 256;
    float*  psum   = (float*)(ws + off); off += 512;
    int*    bucket = (int*)(ws + off);   off += (size_t)EE * SS * 4;
    float*  wts    = (float*)(ws + off); off += (size_t)EE * SS * 4;
    __bf16* xbf    = (__bf16*)(ws + off); off += (size_t)SS * CC * 2;
    __bf16* W1sw   = (__bf16*)(ws + off); off += (size_t)EE * CC * HH * 2;
    __bf16* W3sw   = (__bf16*)(ws + off); off += (size_t)EE * CC * HH * 2;
    __bf16* W2sw   = (__bf16*)(ws + off); off += (size_t)EE * HH * CC * 2;

    const int swzBlocks = (int)(((size_t)EE * CC * HH / 4 + 255) / 256);   // 32768

    moe_init<<<2048, 256, 0, stream>>>(y, counts, fcnt, psum);
    cast_x<<<4096, 256, 0, stream>>>(x, xbf);
    swizzle_w13<<<swzBlocks, 256, 0, stream>>>(W1, W1sw);
    swizzle_w13<<<swzBlocks, 256, 0, stream>>>(W3, W3sw);
    swizzle_w2 <<<swzBlocks, 256, 0, stream>>>(W2, W2sw);
    gate_kernel<<<SS / 8, 256, 0, stream>>>(x, Wg, bg, counts, bucket, wts, fcnt, psum);
    moe_ffn<<<EE * TMAX, 256, 0, stream>>>(xbf, W1sw, b1, W3sw, b3, W2sw, b2,
                                           counts, bucket, wts, y);
    aux_finalize<<<1, 32, 0, stream>>>(fcnt, psum, y + (size_t)SS * CC);
}
